// SubfeatureGroupingAttention_17239998726193
// MI455X (gfx1250) — matrix operand order repaired and checked by reading, not compiled, not run on hardware
//
#include <hip/hip_runtime.h>
#include <hip/hip_bf16.h>

typedef __attribute__((ext_vector_type(16))) _Float16 v16h;
typedef __attribute__((ext_vector_type(8)))  _Float16 v8h;
typedef __attribute__((ext_vector_type(8)))  float    v8f;
typedef __attribute__((ext_vector_type(4)))  float    v4f;
typedef __attribute__((ext_vector_type(4)))  unsigned int v4u;
typedef __attribute__((ext_vector_type(8)))  int      v8i;
typedef __attribute__((ext_vector_type(4)))  int      v4i;

#define DAU 35
#define DLM 196
#define DGZ 2
#define CIN 235
#define DM  128
#define NPOS (32 * 4096)
#define WAVES 4
#define PPW 16            // positions per wave

// packed weight fragment offsets (units: halves; each fragment = 32x16 = 512 halves)
#define OFF_AU  0
#define OFF_LM  (16 * 512)
#define OFF_GZ  (72 * 512)
#define OFF_IN  (80 * 512)
#define OFF_OUT (176 * 512)
#define NFRAG   208

// per-wave LDS layout (bytes). FEAT stride 265 f32: odd stride -> bank-conflict
// free, and wide enough (max k read = 262) that A-fragment builds never need
// bounds checks -- the zeroed B-fragment rows (k >= Kdim) nullify pad lanes.
#define FEAT_STRIDE 265
#define LDS_FEAT 0
#define LDS_TOK  (16 * FEAT_STRIDE * 4)            // 16960
#define LDS_HB   (LDS_TOK + 48 * 128 * 4)          // 41536
#define LDS_OB   (LDS_HB + 48 * 96 * 2)            // 50752
#define LDS_LG   (LDS_OB + 48 * 128 * 2)           // 63040
#define LDS_PER_WAVE_PAD 63488

#define WMMA_F16(acc, a, b) \
  (acc) = __builtin_amdgcn_wmma_f32_16x16x32_f16(false, (a), false, (b), (short)0, (acc), false, false)

// ---------------------------------------------------------------------------
// Pack weights (fp32, row-major [Odim, Kdim]) into f16 WMMA B-fragments.
// Rows with k >= Kdim are zero -- this is what makes unchecked A builds safe.
// ---------------------------------------------------------------------------
__global__ void sga_prep_weights(const float* __restrict__ au_w,
                                 const float* __restrict__ lm_w,
                                 const float* __restrict__ gz_w,
                                 const float* __restrict__ in_w,
                                 const float* __restrict__ out_w,
                                 _Float16* __restrict__ wf) {
  const int f = blockIdx.x;
  const float* W;
  int Kdim, kt = 0, nt = 0;
  if (f < 16)       { W = au_w;  Kdim = DAU; kt = f / 8;          nt = f % 8; }
  else if (f < 72)  { W = lm_w;  Kdim = DLM; kt = (f - 16) / 8;   nt = (f - 16) % 8; }
  else if (f < 80)  { W = gz_w;  Kdim = DGZ; kt = 0;              nt = f - 72; }
  else if (f < 176) { W = in_w;  Kdim = DM;  kt = (f - 80) / 24;  nt = (f - 80) % 24; }
  else              { W = out_w; Kdim = DM;  kt = (f - 176) / 8;  nt = (f - 176) % 8; }
  for (int e = threadIdx.x; e < 512; e += blockDim.x) {
    const int lane = e >> 4, h = e & 15;
    const int n = nt * 16 + (lane & 15);
    const int k = kt * 32 + ((h < 8) ? h : (h + 8)) + ((lane >= 16) ? 8 : 0);
    const float v = (k < Kdim) ? W[n * Kdim + k] : 0.0f;
    wf[f * 512 + e] = (_Float16)v;
  }
}

// A fragment (16-bit, 16x32): lane<16 -> row=lane, K = {h<8 ? h : h+8};
// lane>=16 -> +8. Unconditional scalar build (source rows are padded).
__device__ __forceinline__ v16h build_a_f32(const float* base, int rstride,
                                            int k0, int lane) {
  v16h a;
  const int r  = lane & 15;
  const int kb = (lane & 16) ? 8 : 0;
#pragma unroll
  for (int h = 0; h < 16; ++h) {
    const int k = k0 + kb + ((h < 8) ? h : (h + 8));
    a[h] = (_Float16)base[r * rstride + k];
  }
  return a;
}

// Vectorized builds for 16B-aligned tiles (stride % 4 == 0, k0 % 4 == 0).
__device__ __forceinline__ v16h build_a_f32_v(const float* base, int rstride,
                                              int k0, int lane) {
  const int r  = lane & 15;
  const int kb = (lane & 16) ? 8 : 0;
  const float* p0 = base + r * rstride + k0 + kb;   // K run 0: 8 floats
  const v4f x0 = *(const v4f*)p0;
  const v4f x1 = *(const v4f*)(p0 + 4);
  const v4f y0 = *(const v4f*)(p0 + 16);            // K run 1: +16
  const v4f y1 = *(const v4f*)(p0 + 20);
  v16h a;
#pragma unroll
  for (int c = 0; c < 4; ++c) {
    a[c]      = (_Float16)x0[c];
    a[4 + c]  = (_Float16)x1[c];
    a[8 + c]  = (_Float16)y0[c];
    a[12 + c] = (_Float16)y1[c];
  }
  return a;
}

__device__ __forceinline__ v16h build_a_f16_v(const _Float16* base, int rstride,
                                              int k0, int lane) {
  const int r  = lane & 15;
  const int kb = (lane & 16) ? 8 : 0;
  const _Float16* p0 = base + r * rstride + k0 + kb;
  const v8h lo = *(const v8h*)p0;
  const v8h hi = *(const v8h*)(p0 + 16);
  v16h a;
#pragma unroll
  for (int c = 0; c < 8; ++c) { a[c] = lo[c]; a[8 + c] = hi[c]; }
  return a;
}

__device__ __forceinline__ v16h load_b(const _Float16* wbase, int frag, int lane) {
  return *(const v16h*)(wbase + frag * 512 + lane * 16);
}

// ---------------------------------------------------------------------------
// Main fused kernel: one wave == 16 positions, full pipeline.
// ---------------------------------------------------------------------------
__global__ void sga_main(const float* __restrict__ video,
                         const float* __restrict__ au_b,
                         const float* __restrict__ lm_b,
                         const float* __restrict__ gz_b,
                         const float* __restrict__ in_b,
                         const float* __restrict__ out_b,
                         const float* __restrict__ ln_g,
                         const float* __restrict__ ln_b,
                         const float* __restrict__ w_w,
                         const float* __restrict__ w_b,
                         const _Float16* __restrict__ wf,
                         float* __restrict__ out) {
  extern __shared__ char smem[];
  const int lane = threadIdx.x & 31;
  const int wid  = threadIdx.x >> 5;
  const long pos0 = (long)(blockIdx.x * WAVES + wid) * PPW;

  char* my = smem + wid * LDS_PER_WAVE_PAD;
  float*    feat = (float*)(my + LDS_FEAT);
  float*    tokb = (float*)(my + LDS_TOK);
  _Float16* hb   = (_Float16*)(my + LDS_HB);
  _Float16* ob   = (_Float16*)(my + LDS_OB);
  float*    lg   = (float*)(my + LDS_LG);

  const int col   = lane & 15;
  const int rhalf = (lane & 16) ? 8 : 0;

  __builtin_prefetch(wf, 0, 3);  // keep packed weights warm (global_prefetch)

  // Zero the 30 pad columns so padded A reads are finite (never NaN/Inf);
  // their products are nullified by the zeroed B rows anyway.
  for (int i = lane; i < PPW * (FEAT_STRIDE - CIN); i += 32) {
    const int r = i / (FEAT_STRIDE - CIN), c = i % (FEAT_STRIDE - CIN);
    feat[r * FEAT_STRIDE + CIN + c] = 0.0f;
  }

  // ---- Stage 1: 16x235 f32 tile -> LDS via Tensor Data Mover ----
  // Iterate mode: 16 iterations of a 1x235 row, global step 235 elems,
  // LDS step 265 elems -> rows land at the padded stride.
  const float* src = video + pos0 * CIN;
#if __has_builtin(__builtin_amdgcn_tensor_load_to_lds)
  {
    const unsigned lds_off = (unsigned)(uintptr_t)feat;      // low 32b = LDS byte addr
    const unsigned long long ga = (unsigned long long)(uintptr_t)src;
    // D# group0: count=1 | lds_addr | global_addr[56:0] | type=2
    const v4u g0 = { 1u, lds_off, (unsigned)(ga & 0xffffffffu),
                     (unsigned)((ga >> 32) & 0x01ffffffu) | (2u << 30) };
    // D# group1: data_size=4B, iterate_enable; tensor_dim0=235,
    // tensor_dim1=NPOS; tile_dim0=235, tile_dim1=1; tensor_dim0_stride=235
    const v8i g1 = { (int)((2u << 16) | (1u << 19)),
                     (int)(((unsigned)CIN & 0xffffu) << 16),
                     0,
                     (int)(((unsigned)NPOS >> 16) | ((unsigned)CIN << 16)),
                     1,
                     (int)CIN,
                     0, 0 };
    // D# group2 (iterate mode): lds_addr_increment=265, global_addr_increment=235,
    // iterate_count=15 (-> 16 iterations)
    const v4i g2 = { 0, FEAT_STRIDE, CIN, (int)((PPW - 1) << 16) };
    const v4i g3 = { 0, 0, 0, 0 };
#if defined(__clang_major__) && (__clang_major__ >= 23)
    const v8i gz8 = { 0, 0, 0, 0, 0, 0, 0, 0 };
    __builtin_amdgcn_tensor_load_to_lds(g0, g1, g2, g3, gz8, 0);
#else
    __builtin_amdgcn_tensor_load_to_lds(g0, g1, g2, g3, 0);
#endif
  }
#if __has_builtin(__builtin_amdgcn_s_wait_tensorcnt)
  __builtin_amdgcn_s_wait_tensorcnt(0);
#else
  asm volatile("s_wait_tensorcnt 0" ::: "memory");
#endif
#else
  for (int i = lane; i < PPW * CIN; i += 32) {
    const int r = i / CIN, c = i - r * CIN;
    feat[r * FEAT_STRIDE + c] = src[i];
  }
  asm volatile("s_wait_dscnt 0" ::: "memory");
#endif

  // ---- Stage 2: three input projections -> tok (48x128 f32 in LDS) ----
  {  // AU: K=35 (2 k-tiles)
    v8f acc[8] = {};
#pragma unroll
    for (int kt = 0; kt < 2; ++kt) {
      const v16h a = build_a_f32(feat, FEAT_STRIDE, kt * 32, lane);
#pragma unroll
      for (int nt = 0; nt < 8; ++nt)
        WMMA_F16(acc[nt], a, load_b(wf + OFF_AU, kt * 8 + nt, lane));
    }
#pragma unroll
    for (int nt = 0; nt < 8; ++nt) {
      const float bias = au_b[nt * 16 + col];
#pragma unroll
      for (int rr = 0; rr < 8; ++rr)
        tokb[(rhalf + rr) * DM + nt * 16 + col] = acc[nt][rr] + bias;
    }
  }
  {  // LM: K=196 (7 k-tiles), feature cols 35..230
    v8f acc[8] = {};
#pragma unroll
    for (int kt = 0; kt < 7; ++kt) {
      const v16h a = build_a_f32(feat, FEAT_STRIDE, DAU + kt * 32, lane);
#pragma unroll
      for (int nt = 0; nt < 8; ++nt)
        WMMA_F16(acc[nt], a, load_b(wf + OFF_LM, kt * 8 + nt, lane));
    }
#pragma unroll
    for (int nt = 0; nt < 8; ++nt) {
      const float bias = lm_b[nt * 16 + col];
#pragma unroll
      for (int rr = 0; rr < 8; ++rr)
        tokb[(16 + rhalf + rr) * DM + nt * 16 + col] = acc[nt][rr] + bias;
    }
  }
  {  // GZ: K=2 (1 k-tile), feature cols 231..232
    v8f acc[8] = {};
    const v16h a = build_a_f32(feat, FEAT_STRIDE, DAU + DLM, lane);
#pragma unroll
    for (int nt = 0; nt < 8; ++nt)
      WMMA_F16(acc[nt], a, load_b(wf + OFF_GZ, nt, lane));
#pragma unroll
    for (int nt = 0; nt < 8; ++nt) {
      const float bias = gz_b[nt * 16 + col];
#pragma unroll
      for (int rr = 0; rr < 8; ++rr)
        tokb[(32 + rhalf + rr) * DM + nt * 16 + col] = acc[nt][rr] + bias;
    }
  }
  asm volatile("s_wait_dscnt 0" ::: "memory");

  // ---- Stage 3: per-head QKV (WMMA) + tiny 3-token attention (VALU) ----
#pragma unroll 1
  for (int h = 0; h < 4; ++h) {
#pragma unroll
    for (int t = 0; t < 3; ++t) {
      v8f acc[6] = {};
#pragma unroll
      for (int kt = 0; kt < 4; ++kt) {
        const v16h a = build_a_f32_v(tokb + t * 16 * DM, DM, kt * 32, lane);
#pragma unroll
        for (int u = 0; u < 6; ++u) {
          const int nn = (u >> 1) * 8 + 2 * h + (u & 1);
          WMMA_F16(acc[u], a, load_b(wf + OFF_IN, kt * 24 + nn, lane));
        }
      }
#pragma unroll
      for (int u = 0; u < 6; ++u) {
        const int g = u >> 1, j = u & 1;
        const int nn = g * 8 + 2 * h + j;
        const float bias = in_b[nn * 16 + col];
        const int hcol = g * 32 + j * 16 + col;
#pragma unroll
        for (int rr = 0; rr < 8; ++rr)
          hb[(t * 16 + rhalf + rr) * 96 + hcol] = (_Float16)(acc[u][rr] + bias);
      }
    }
    asm volatile("s_wait_dscnt 0" ::: "memory");

    if (lane < 16) {  // one position per lane: 3x3 attention over d=32
      const int p = lane;
      float s[3][3];
#pragma unroll
      for (int i = 0; i < 3; ++i)
#pragma unroll
        for (int j = 0; j < 3; ++j) s[i][j] = 0.0f;
#pragma unroll
      for (int dq = 0; dq < 4; ++dq) {
        v8h qv[3], kv[3];
#pragma unroll
        for (int i = 0; i < 3; ++i) {
          qv[i] = *(const v8h*)&hb[(i * 16 + p) * 96 + dq * 8];
          kv[i] = *(const v8h*)&hb[(i * 16 + p) * 96 + 32 + dq * 8];
        }
#pragma unroll
        for (int e = 0; e < 8; ++e)
#pragma unroll
          for (int i = 0; i < 3; ++i)
#pragma unroll
            for (int j = 0; j < 3; ++j)
              s[i][j] += (float)qv[i][e] * (float)kv[j][e];
      }
      const float scale = 0.17677669529663687f;  // 1/sqrt(32)
      float aw[3][3];
#pragma unroll
      for (int i = 0; i < 3; ++i) {
        const float m = fmaxf(s[i][0], fmaxf(s[i][1], s[i][2])) * scale;
        const float e0 = __expf(s[i][0] * scale - m);
        const float e1 = __expf(s[i][1] * scale - m);
        const float e2 = __expf(s[i][2] * scale - m);
        const float inv = 1.0f / (e0 + e1 + e2);
        aw[i][0] = e0 * inv; aw[i][1] = e1 * inv; aw[i][2] = e2 * inv;
      }
#pragma unroll
      for (int dq = 0; dq < 4; ++dq) {
        v8h vv[3];
#pragma unroll
        for (int j = 0; j < 3; ++j)
          vv[j] = *(const v8h*)&hb[(j * 16 + p) * 96 + 64 + dq * 8];
#pragma unroll
        for (int i = 0; i < 3; ++i) {
          v8h o;
#pragma unroll
          for (int e = 0; e < 8; ++e)
            o[e] = (_Float16)(aw[i][0] * (float)vv[0][e] +
                              aw[i][1] * (float)vv[1][e] +
                              aw[i][2] * (float)vv[2][e]);
          *(v8h*)&ob[(i * 16 + p) * DM + h * 32 + dq * 8] = o;
        }
      }
    }
    asm volatile("s_wait_dscnt 0" ::: "memory");
  }

  // ---- Stage 4: output projection + residual: x = o @ out_w.T + out_b + tok ----
#pragma unroll 1
  for (int t = 0; t < 3; ++t) {
    v8f acc[8] = {};
#pragma unroll
    for (int kt = 0; kt < 4; ++kt) {
      const v16h a = build_a_f16_v(ob + t * 16 * DM, DM, kt * 32, lane);
#pragma unroll
      for (int nt = 0; nt < 8; ++nt)
        WMMA_F16(acc[nt], a, load_b(wf + OFF_OUT, kt * 8 + nt, lane));
    }
#pragma unroll
    for (int nt = 0; nt < 8; ++nt) {
      const float bias = out_b[nt * 16 + col];
#pragma unroll
      for (int rr = 0; rr < 8; ++rr) {
        const int idx = (t * 16 + rhalf + rr) * DM + nt * 16 + col;
        tokb[idx] += acc[nt][rr] + bias;
      }
    }
  }
  asm volatile("s_wait_dscnt 0" ::: "memory");

  // ---- Stage 5: LayerNorm + gating logits (VALU, float4) ----
  for (int rw = lane; rw < 48; rw += 32) {
    v4f* xr = (v4f*)(tokb + rw * DM);
    float mu = 0.0f;
    for (int q = 0; q < 32; ++q) {
      const v4f t = xr[q];
      mu += t[0] + t[1] + t[2] + t[3];
    }
    mu *= (1.0f / DM);
    float var = 0.0f;
    for (int q = 0; q < 32; ++q) {
      const v4f t = xr[q];
#pragma unroll
      for (int c = 0; c < 4; ++c) { const float df = t[c] - mu; var += df * df; }
    }
    var *= (1.0f / DM);
    const float inv = rsqrtf(var + 1e-5f);
    float logit = w_b[0];
    for (int q = 0; q < 32; ++q) {
      const v4f t  = xr[q];
      const v4f g  = *(const v4f*)(ln_g + q * 4);
      const v4f bb = *(const v4f*)(ln_b + q * 4);
      const v4f wv = *(const v4f*)(w_w + q * 4);
      v4f xn;
#pragma unroll
      for (int c = 0; c < 4; ++c) {
        xn[c] = (t[c] - mu) * inv * g[c] + bb[c];
        logit += xn[c] * wv[c];
      }
      xr[q] = xn;
    }
    lg[rw] = logit;
  }
  asm volatile("s_wait_dscnt 0" ::: "memory");

  // ---- Stage 6: token softmax + fuse, 128-bit coalesced stores ----
  for (int i = lane; i < PPW * DM / 4; i += 32) {
    const int p = i >> 5, q = i & 31;
    const float l0 = lg[p], l1 = lg[16 + p], l2 = lg[32 + p];
    const float m = fmaxf(l0, fmaxf(l1, l2));
    const float e0 = __expf(l0 - m), e1 = __expf(l1 - m), e2 = __expf(l2 - m);
    const float inv = 1.0f / (e0 + e1 + e2);
    const v4f x0 = ((const v4f*)(tokb + p * DM))[q];
    const v4f x1 = ((const v4f*)(tokb + (16 + p) * DM))[q];
    const v4f x2 = ((const v4f*)(tokb + (32 + p) * DM))[q];
    v4f r;
#pragma unroll
    for (int c = 0; c < 4; ++c)
      r[c] = (e0 * x0[c] + e1 * x1[c] + e2 * x2[c]) * inv;
    ((v4f*)(out + (pos0 + p) * DM))[q] = r;
  }
}

extern "C" void kernel_launch(void* const* d_in, const int* in_sizes, int n_in,
                              void* d_out, int out_size, void* d_ws, size_t ws_size,
                              hipStream_t stream) {
  const float* video = (const float*)d_in[0];
  const float* au_w  = (const float*)d_in[1];
  const float* au_bv = (const float*)d_in[2];
  const float* lm_w  = (const float*)d_in[3];
  const float* lm_bv = (const float*)d_in[4];
  const float* gz_w  = (const float*)d_in[5];
  const float* gz_bv = (const float*)d_in[6];
  const float* in_w  = (const float*)d_in[7];
  const float* in_bv = (const float*)d_in[8];
  const float* out_w = (const float*)d_in[9];
  const float* out_bv= (const float*)d_in[10];
  const float* ln_g  = (const float*)d_in[11];
  const float* ln_bv = (const float*)d_in[12];
  const float* w_w   = (const float*)d_in[13];
  const float* w_bv  = (const float*)d_in[14];
  _Float16* wf = (_Float16*)d_ws;
  float* outp  = (float*)d_out;

  hipLaunchKernelGGL(sga_prep_weights, dim3(NFRAG), dim3(256), 0, stream,
                     au_w, lm_w, gz_w, in_w, out_w, wf);

  const int blocks = NPOS / (WAVES * PPW);               // 2048
  const size_t smem = (size_t)WAVES * LDS_PER_WAVE_PAD;  // ~248 KB (<320 KB WGP)
  hipLaunchKernelGGL(sga_main, dim3(blocks), dim3(WAVES * 32), smem, stream,
                     video, au_bv, lm_bv, gz_bv, in_bv, out_bv,
                     ln_g, ln_bv, w_w, w_bv, (const _Float16*)wf, outp);
}
